// Codebook_36747740184891
// MI455X (gfx1250) — matrix operand order repaired and checked by reading, not compiled, not run on hardware
//
#include <hip/hip_runtime.h>
#include <hip/hip_bf16.h>

typedef __attribute__((ext_vector_type(16))) __bf16 v16bf;
typedef __attribute__((ext_vector_type(8)))  __bf16 v8bf;
typedef __attribute__((ext_vector_type(8)))  float  v8f;

#define D_DIM 256
#define KSTEPS 8   /* 256 / 32 */

// ---------------------------------------------------------------------------
// Kernel 0: gather sub-codebook rows, split f32 -> bf16 hi/lo, compute ||c||^2.
// Pad rows [K, KP) get zero data and c2 = 3e38 so they never win the argmin.
// One wave (32 lanes) per code row; lane handles 8 consecutive dims.
// ---------------------------------------------------------------------------
__global__ void __launch_bounds__(32)
prep_codes(const float* __restrict__ codebook,
           const int* __restrict__ ridx,
           int K,
           __bf16* __restrict__ subHi,
           __bf16* __restrict__ subLo,
           float* __restrict__ c2)
{
    int k    = blockIdx.x;
    int lane = threadIdx.x;
    int d0   = lane * 8;
    float sum = 0.0f;

    if (k < K) {
        int row = ridx[k];
        const float4* src = reinterpret_cast<const float4*>(codebook + (size_t)row * D_DIM + d0);
        float4 f0 = src[0];
        float4 f1 = src[1];
        float f[8] = { f0.x, f0.y, f0.z, f0.w, f1.x, f1.y, f1.z, f1.w };
        v8bf h, l;
        #pragma unroll
        for (int j = 0; j < 8; ++j) {
            float v  = f[j];
            __bf16 hb = (__bf16)v;
            __bf16 lb = (__bf16)(v - (float)hb);
            h[j] = hb;
            l[j] = lb;
            sum += v * v;
        }
        *reinterpret_cast<v8bf*>(subHi + (size_t)k * D_DIM + d0) = h;
        *reinterpret_cast<v8bf*>(subLo + (size_t)k * D_DIM + d0) = l;
    } else {
        v8bf z = {};
        *reinterpret_cast<v8bf*>(subHi + (size_t)k * D_DIM + d0) = z;
        *reinterpret_cast<v8bf*>(subLo + (size_t)k * D_DIM + d0) = z;
    }

    #pragma unroll
    for (int off = 16; off >= 1; off >>= 1)
        sum += __shfl_xor(sum, off, 32);
    if (lane == 0)
        c2[k] = (k < K) ? sum : 3.0e38f;
}

// ---------------------------------------------------------------------------
// Kernel 1: fused GEMM + argmin. One wave per 16 tokens.
// A (16x256) held in registers as bf16 hi/lo WMMA fragments (contiguous k
// order: half hi holds K = hi*16..+16); B streamed from the
// gathered sub-codebook (fits in L2). Hi/lo error-compensated bf16 product
// (~f32 accuracy) is split across THREE independent accumulator chains
// (HH, HL, LH) so the matrix pipe can keep multiple WMMAs in flight instead
// of serializing a single D->C chain.
// Epilogue: per-lane min of (c2[n] - 2*dot), shfl reduction per 16-lane half.
// ---------------------------------------------------------------------------
__global__ void __launch_bounds__(32)
argmin_gemm(const float* __restrict__ x,
            const __bf16* __restrict__ subHi,
            const __bf16* __restrict__ subLo,
            const float* __restrict__ c2,
            int ntiles,
            int* __restrict__ lidx)
{
    int lane  = threadIdx.x;
    int mr    = lane & 15;     // row (A) / col (B,C) within tile
    int hi    = lane >> 4;     // which half-wave
    int token = blockIdx.x * 16 + mr;

    // ---- load + split A fragments once (contiguous k = 16*hi + j layout) ----
    v16bf aHi[KSTEPS], aLo[KSTEPS];
    #pragma unroll
    for (int ks = 0; ks < KSTEPS; ++ks) {
        const float* p0 = x + (size_t)token * D_DIM + ks * 32 + hi * 16;
        float4 c0a = *reinterpret_cast<const float4*>(p0);
        float4 c0b = *reinterpret_cast<const float4*>(p0 + 4);
        float4 c1a = *reinterpret_cast<const float4*>(p0 + 8);
        float4 c1b = *reinterpret_cast<const float4*>(p0 + 12);
        float f[16] = { c0a.x, c0a.y, c0a.z, c0a.w, c0b.x, c0b.y, c0b.z, c0b.w,
                        c1a.x, c1a.y, c1a.z, c1a.w, c1b.x, c1b.y, c1b.z, c1b.w };
        #pragma unroll
        for (int j = 0; j < 16; ++j) {
            __bf16 hb = (__bf16)f[j];
            aHi[ks][j] = hb;
            aLo[ks][j] = (__bf16)(f[j] - (float)hb);
        }
    }

    float mv[8];
    int   mi[8];
    #pragma unroll
    for (int r = 0; r < 8; ++r) { mv[r] = 3.4e38f; mi[r] = 0; }

    for (int nt = 0; nt < ntiles; ++nt) {
        int n = nt * 16 + mr;                     // this lane's column (code id)
        const __bf16* bh = subHi + (size_t)n * D_DIM + hi * 16;
        const __bf16* bl = subLo + (size_t)n * D_DIM + hi * 16;
        if (nt + 1 < ntiles) {
            __builtin_prefetch(bh + (size_t)16 * D_DIM, 0, 1);
            __builtin_prefetch(bl + (size_t)16 * D_DIM, 0, 1);
        }
        // three independent accumulator chains -> ILP in the matrix pipe
        v8f accHH = {};
        v8f accHL = {};
        v8f accLH = {};
        #pragma unroll
        for (int ks = 0; ks < KSTEPS; ++ks) {
            v16bf bHi = *reinterpret_cast<const v16bf*>(bh + ks * 32);
            v16bf bLo = *reinterpret_cast<const v16bf*>(bl + ks * 32);
            accHH = __builtin_amdgcn_wmma_f32_16x16x32_bf16(false, aHi[ks], false, bHi,
                                                            (short)0, accHH, false, false);
            accHL = __builtin_amdgcn_wmma_f32_16x16x32_bf16(false, aHi[ks], false, bLo,
                                                            (short)0, accHL, false, false);
            accLH = __builtin_amdgcn_wmma_f32_16x16x32_bf16(false, aLo[ks], false, bHi,
                                                            (short)0, accLH, false, false);
        }
        float cc = c2[n];
        #pragma unroll
        for (int r = 0; r < 8; ++r) {
            float dot = accHH[r] + (accHL[r] + accLH[r]);
            float v = cc - 2.0f * dot;            // d2 minus row-constant x2
            if (v < mv[r]) { mv[r] = v; mi[r] = n; }
        }
    }

    // reduce across the 16 lanes of each half (rows r / r+8)
    #pragma unroll
    for (int off = 1; off <= 8; off <<= 1) {
        #pragma unroll
        for (int r = 0; r < 8; ++r) {
            float ov = __shfl_xor(mv[r], off, 32);
            int   oi = __shfl_xor(mi[r], off, 32);
            if (ov < mv[r] || (ov == mv[r] && oi < mi[r])) { mv[r] = ov; mi[r] = oi; }
        }
    }
    if (mr == 0) {
        int rowBase = blockIdx.x * 16 + hi * 8;
        #pragma unroll
        for (int r = 0; r < 8; ++r)
            lidx[rowBase + r] = mi[r];
    }
}

// ---------------------------------------------------------------------------
// Kernel 2: gather winning codebook rows to output + per-token squared error.
// Deterministic wave reduction (no float atomics -> bit-identical replays).
// ---------------------------------------------------------------------------
__global__ void __launch_bounds__(32)
gather_loss(const float* __restrict__ x,
            const float* __restrict__ codebook,
            const int* __restrict__ ridx,
            const int* __restrict__ lidx,
            float* __restrict__ outTok,
            float* __restrict__ outIdx,
            float* __restrict__ partials)
{
    int t    = blockIdx.x;
    int lane = threadIdx.x;
    int code = ridx[lidx[t]];
    const float* crow = codebook + (size_t)code * D_DIM;
    const float* xrow = x + (size_t)t * D_DIM;
    float*       orow = outTok + (size_t)t * D_DIM;

    float s = 0.0f;
    #pragma unroll
    for (int j = 0; j < 2; ++j) {                 // 32 lanes * 4 * 2 = 256 dims
        int d = lane * 4 + j * 128;
        float4 cv = *reinterpret_cast<const float4*>(crow + d);
        float4 xv = *reinterpret_cast<const float4*>(xrow + d);
        *reinterpret_cast<float4*>(orow + d) = cv;
        float dx = cv.x - xv.x, dy = cv.y - xv.y;
        float dz = cv.z - xv.z, dw = cv.w - xv.w;
        s += dx * dx + dy * dy + dz * dz + dw * dw;
    }
    #pragma unroll
    for (int off = 16; off >= 1; off >>= 1)
        s += __shfl_xor(s, off, 32);
    if (lane == 0) {
        partials[t] = s;
        outIdx[t]   = (float)code;
    }
}

// ---------------------------------------------------------------------------
// Kernel 3: deterministic final reduction of per-token partials -> mean loss.
// ---------------------------------------------------------------------------
__global__ void __launch_bounds__(256)
final_loss(const float* __restrict__ partials, int n, float invCount,
           float* __restrict__ outLoss)
{
    __shared__ float sm[256];
    int tid = threadIdx.x;
    float s = 0.0f;
    for (int i = tid; i < n; i += 256)
        s += partials[i];
    sm[tid] = s;
    __syncthreads();
    for (int st = 128; st > 0; st >>= 1) {
        if (tid < st) sm[tid] += sm[tid + st];
        __syncthreads();
    }
    if (tid == 0)
        outLoss[0] = sm[0] * invCount;
}

// ---------------------------------------------------------------------------
extern "C" void kernel_launch(void* const* d_in, const int* in_sizes, int n_in,
                              void* d_out, int out_size, void* d_ws, size_t ws_size,
                              hipStream_t stream)
{
    const float* x        = (const float*)d_in[0];
    const float* codebook = (const float*)d_in[1];
    const int*   ridx     = (const int*)d_in[2];

    int N      = in_sizes[0] / D_DIM;       // 16384 tokens
    int K      = in_sizes[2];               // 3686 active codes
    int KP     = (K + 15) & ~15;            // padded to 16
    int ntiles = KP / 16;

    // workspace carve-out (~3.9 MB)
    char*  ws  = (char*)d_ws;
    size_t off = 0;
    auto alloc = [&](size_t bytes) -> char* {
        char* p = ws + off;
        off = (off + bytes + 255) & ~(size_t)255;
        return p;
    };
    __bf16* subHi    = (__bf16*)alloc((size_t)KP * D_DIM * sizeof(__bf16));
    __bf16* subLo    = (__bf16*)alloc((size_t)KP * D_DIM * sizeof(__bf16));
    float*  c2       = (float*)alloc((size_t)KP * sizeof(float));
    int*    lidx     = (int*)alloc((size_t)N * sizeof(int));
    float*  partials = (float*)alloc((size_t)N * sizeof(float));

    float* outTok  = (float*)d_out;                 // [N*256] closest tokens
    float* outLoss = outTok + (size_t)N * D_DIM;    // [1] rounding loss
    float* outIdx  = outLoss + 1;                   // [N] indices (as float)

    prep_codes<<<KP, 32, 0, stream>>>(codebook, ridx, K, subHi, subLo, c2);
    argmin_gemm<<<N / 16, 32, 0, stream>>>(x, subHi, subLo, c2, ntiles, lidx);
    gather_loss<<<N, 32, 0, stream>>>(x, codebook, ridx, lidx, outTok, outIdx, partials);
    final_loss<<<1, 256, 0, stream>>>(partials, N, 1.0f / ((float)N * (float)D_DIM), outLoss);
}
